// gen_heter_Emb_75625784148229
// MI455X (gfx1250) — matrix operand order repaired and checked by reading, not compiled, not run on hardware
//
#include <hip/hip_runtime.h>
#include <hip/hip_bf16.h>

// ---------------------------------------------------------------------------
// CDNA5 (gfx1250, wave32) implementation.
//
// Heavy math: V_WMMA_F32_16X16X32_BF16. Each workgroup (256 threads = 8
// waves) computes a 128x128 output tile; each wave owns a 64x32 sub-tile
// = 4x2 WMMA accumulators. f32 inputs/weights are converted to bf16 while
// staged into LDS (no extra HBM traffic). Up to three A-segments emulate
// jnp.concatenate along K without materializing it. Global loads for the
// next K-block are issued into registers while the current block's DS
// fragment loads + WMMAs execute (register double-buffering), and the W
// tile two K-steps ahead is prefetched into L2 via global_prefetch_b8.
// ---------------------------------------------------------------------------

typedef __attribute__((ext_vector_type(16))) __bf16        v16bf;
typedef __attribute__((ext_vector_type(8)))  float         v8f;
typedef __attribute__((ext_vector_type(8)))  unsigned int  v8u;

#define TILE_M 128
#define TILE_N 128
#define TILE_K 32
#define LDS_STRIDE 34   // halves; even -> 32-bit pair reads stay 4B aligned

// Load one 128x32 A sub-block (f32) into 16 registers per thread.
// Linear element i of thread t maps to row=(t*16+i)>>5, k=(t*16+i)&31 in
// BOTH paths, so a single store function works for either.
__device__ __forceinline__
void load_A_regs(const float* __restrict__ As, int tileM, int kloc, int ld,
                 int t, float (&r)[16])
{
    if (((ld & 3) == 0) && (kloc + TILE_K <= ld)) {
#pragma unroll
        for (int ii = 0; ii < 4; ++ii) {
            const int idx4 = t * 4 + ii;
            const int row  = idx4 >> 3;          // 0..127
            const int kq   = (idx4 & 7) << 2;    // 0..28
            const float4 v = *(const float4*)(As + (size_t)(tileM + row) * ld + kloc + kq);
            r[ii * 4 + 0] = v.x; r[ii * 4 + 1] = v.y;
            r[ii * 4 + 2] = v.z; r[ii * 4 + 3] = v.w;
        }
    } else {
#pragma unroll
        for (int i = 0; i < 16; ++i) {
            const int idx = t * 16 + i;
            const int row = idx >> 5;
            const int k   = idx & 31;
            r[i] = (kloc + k < ld) ? As[(size_t)(tileM + row) * ld + kloc + k] : 0.f;
        }
    }
}

// Load one 32x128 W sub-block (f32) into 16 registers per thread.
// Linear element i of thread t maps to k=(t*16+i)>>7, col=(t*16+i)&127.
__device__ __forceinline__
void load_B_regs(const float* __restrict__ W, int kb, int Kt, int tileN, int N,
                 int t, float (&r)[16])
{
    if (kb + TILE_K <= Kt) {   // N always multiple of 4
#pragma unroll
        for (int ii = 0; ii < 4; ++ii) {
            const int idx4 = t * 4 + ii;
            const int rr   = idx4 >> 5;          // 0..31
            const int c    = (idx4 & 31) << 2;   // 0..124
            float4 v = make_float4(0.f, 0.f, 0.f, 0.f);
            if (tileN + c < N)
                v = *(const float4*)(W + (size_t)(kb + rr) * N + tileN + c);
            r[ii * 4 + 0] = v.x; r[ii * 4 + 1] = v.y;
            r[ii * 4 + 2] = v.z; r[ii * 4 + 3] = v.w;
        }
    } else {
#pragma unroll
        for (int i = 0; i < 16; ++i) {
            const int idx = t * 16 + i;
            const int rr  = idx >> 7;
            const int c   = idx & 127;
            r[i] = ((kb + rr < Kt) && (tileN + c < N))
                       ? W[(size_t)(kb + rr) * N + tileN + c] : 0.f;
        }
    }
}

__global__ __launch_bounds__(256)
void gemm_bias_act_kernel(const float* __restrict__ A0,
                          const float* __restrict__ A1,
                          const float* __restrict__ A2,
                          int K0, int K1, int K2,
                          const float* __restrict__ W,     // (K0+K1+K2) x N row-major
                          const float* __restrict__ bias,  // N
                          float* __restrict__ out,         // M x N
                          int N, int doRelu, int M)
{
    __shared__ __bf16 lA[TILE_M][LDS_STRIDE];   // A tile: 128 rows x 32 k
    __shared__ __bf16 lB[TILE_N][LDS_STRIDE];   // W tile transposed: 128 cols x 32 k

    const int t    = threadIdx.x;
    const int lane = t & 31;
    const int w    = t >> 5;          // wave id 0..7
    const int wr   = w >> 2;          // wave row 0..1  (64 rows each)
    const int wc   = w & 3;           // wave col 0..3  (32 cols each)

    const int tileM = blockIdx.x * TILE_M;
    const int tileN = blockIdx.y * TILE_N;
    const int Kt    = K0 + K1 + K2;

    v8f acc[4][2];
#pragma unroll
    for (int mr = 0; mr < 4; ++mr)
#pragma unroll
        for (int nc = 0; nc < 2; ++nc)
            acc[mr][nc] = (v8f){0.f,0.f,0.f,0.f,0.f,0.f,0.f,0.f};

    const int lrow   = lane & 15;
    const int kbaseA = (lane >> 4) * 8;    // A frag: lanes16-31 start at K=8
    const int kbaseB = (lane >> 4) * 8;    // B frag: lanes16-31 start at K=8

    // Segment chooser for a given K block base.
    auto pick = [&](int kb, const float*& As, int& kloc, int& ld) {
        if (kb < K0)           { As = A0; kloc = kb;           ld = K0; }
        else if (kb < K0 + K1) { As = A1; kloc = kb - K0;      ld = K1; }
        else                   { As = A2; kloc = kb - K0 - K1; ld = K2; }
    };

    float rA[16], rB[16];
    {   // prologue: fetch first K block into registers
        const float* As; int kloc, ld;
        pick(0, As, kloc, ld);
        load_A_regs(As, tileM, kloc, ld, t, rA);
        load_B_regs(W, 0, Kt, tileN, N, t, rB);
    }

    for (int kb = 0; kb < Kt; kb += TILE_K) {
        __syncthreads();   // all fragment reads of the previous block done

        // ------- commit the registered block to LDS (f32 -> bf16) --------
#pragma unroll
        for (int i = 0; i < 16; ++i) {
            const int idx = t * 16 + i;
            lA[idx >> 5][idx & 31] = (__bf16)rA[i];
        }
#pragma unroll
        for (int i = 0; i < 16; ++i) {
            const int idx = t * 16 + i;
            lB[idx & 127][idx >> 7] = (__bf16)rB[i];
        }

        __syncthreads();   // tile visible to all waves

        // ------- kick off next block's global loads (overlap with WMMA) --
        if (kb + TILE_K < Kt) {
            const float* As; int kloc, ld;
            pick(kb + TILE_K, As, kloc, ld);
            load_A_regs(As, tileM, kloc, ld, t, rA);
            load_B_regs(W, kb + TILE_K, Kt, tileN, N, t, rB);
        }

        // ------- L2 prefetch of the W tile two K-steps ahead -------------
        if (kb + 2 * TILE_K < Kt) {
            const float* pf = W + (size_t)(kb + 2 * TILE_K + (t >> 3)) * N
                                + tileN + ((t & 7) << 4);
            __builtin_prefetch(pf, 0, 1);   // global_prefetch_b8 (speculative-safe)
        }

        // ------------------- build fragments from LDS --------------------
        v16bf afrag[4];
#pragma unroll
        for (int mr = 0; mr < 4; ++mr) {
            const int row = wr * 64 + mr * 16 + lrow;
            v8u raw;
#pragma unroll
            for (int p = 0; p < 8; ++p) {
                const int kk = (p < 4) ? (kbaseA + 2 * p)
                                       : (kbaseA + 16 + 2 * (p - 4));
                raw[p] = *(const unsigned int*)&lA[row][kk];  // 4B-aligned pair
            }
            afrag[mr] = __builtin_bit_cast(v16bf, raw);
        }

        v16bf bfrag[2];
#pragma unroll
        for (int nc = 0; nc < 2; ++nc) {
            const int col = wc * 32 + nc * 16 + lrow;
            v8u raw;
#pragma unroll
            for (int p = 0; p < 8; ++p) {
                const int kk = (p < 4) ? (kbaseB + 2 * p)
                                       : (kbaseB + 16 + 2 * (p - 4));
                raw[p] = *(const unsigned int*)&lB[col][kk];
            }
            bfrag[nc] = __builtin_bit_cast(v16bf, raw);
        }

        // --------------------------- 8 WMMAs -----------------------------
#pragma unroll
        for (int mr = 0; mr < 4; ++mr)
#pragma unroll
            for (int nc = 0; nc < 2; ++nc)
                acc[mr][nc] = __builtin_amdgcn_wmma_f32_16x16x32_bf16(
                    /*neg_a=*/false, afrag[mr],
                    /*neg_b=*/false, bfrag[nc],
                    /*c_mod=*/(short)0, acc[mr][nc],
                    /*reuse_a=*/false, /*reuse_b=*/false);
    }

    // ------------------------- bias + relu + store ------------------------
#pragma unroll
    for (int mr = 0; mr < 4; ++mr) {
#pragma unroll
        for (int nc = 0; nc < 2; ++nc) {
            const int col = tileN + wc * 32 + nc * 16 + (lane & 15);
            if (col >= N) continue;
            const float b = bias[col];
#pragma unroll
            for (int r = 0; r < 8; ++r) {
                const int row = tileM + wr * 64 + mr * 16 + ((lane < 16) ? r : r + 8);
                if (row < M) {
                    float v = acc[mr][nc][r] + b;
                    if (doRelu) v = fmaxf(v, 0.f);
                    out[(size_t)row * N + col] = v;
                }
            }
        }
    }
}

// ---------------------------------------------------------------------------
// Per-edge geometric features:
//   pos[e]  = [ci-cj (3), |ci-cj| (1), (cj-ci)/|ci-cj| (3)]   -> [E,7]
//   geom[e] = log(lwhV[i]/lwhV[j])                            -> [E,4]
// Edge e -> (i, j): i = e/(N-1); j skips the diagonal.
// ---------------------------------------------------------------------------
__global__ __launch_bounds__(256)
void edge_features_kernel(const float* __restrict__ centroid,  // N x 3
                          const float* __restrict__ lwhV,      // N x 4
                          float* __restrict__ posf,            // E x 7
                          float* __restrict__ geomf,           // E x 4
                          int Nn, int E)
{
    const int e = blockIdx.x * blockDim.x + threadIdx.x;
    if (e >= E) return;
    const int i = e / (Nn - 1);
    const int r = e - i * (Nn - 1);
    const int j = (r < i) ? r : r + 1;

    const float cix = centroid[i * 3 + 0], ciy = centroid[i * 3 + 1], ciz = centroid[i * 3 + 2];
    const float cjx = centroid[j * 3 + 0], cjy = centroid[j * 3 + 1], cjz = centroid[j * 3 + 2];
    const float ox = cix - cjx, oy = ciy - cjy, oz = ciz - cjz;
    const float d   = sqrtf(ox * ox + oy * oy + oz * oz);
    const float inv = 1.0f / d;

    float* p = posf + (size_t)e * 7;
    p[0] = ox;  p[1] = oy;  p[2] = oz;
    p[3] = d;
    p[4] = -ox * inv;  p[5] = -oy * inv;  p[6] = -oz * inv;

    float* g = geomf + (size_t)e * 4;
#pragma unroll
    for (int c = 0; c < 4; ++c)
        g[c] = logf(lwhV[i * 4 + c] / lwhV[j * 4 + c]);
}

// ---------------------------------------------------------------------------
// Host-side orchestration (graph-capture safe: only kernel launches).
// Input order: 0 insnum, 1 pred_codes, 2 bboxes, 3 lwhV, 4 centroid,
// 5 sW1, 6 sb1, 7 sW2, 8 sb2, 9 pW1, 10 pb1, 11 pW2, 12 pb2,
// 13 cW1, 14 cb1, 15 cW2, 16 cb2, 17 lW1, 18 lb1, 19 lW2, 20 lb2,
// 21 posW1, 22 posb1, 23 posW2, 24 posb2, 25 posW3, 26 posb3,
// 27 gW1, 28 gb1, 29 gW2, 30 gb2, 31 gW3, 32 gb3.
// Output: [link, support, proximity, comp] each E x 512, concatenated.
// ---------------------------------------------------------------------------
extern "C" void kernel_launch(void* const* d_in, const int* in_sizes, int n_in,
                              void* d_out, int out_size, void* d_ws, size_t ws_size,
                              hipStream_t stream)
{
    (void)n_in; (void)out_size; (void)ws_size;

    const float* pred     = (const float*)d_in[1];
    const float* lwhV     = (const float*)d_in[3];
    const float* centroid = (const float*)d_in[4];
    const float* sW1 = (const float*)d_in[5],  *sb1 = (const float*)d_in[6];
    const float* sW2 = (const float*)d_in[7],  *sb2 = (const float*)d_in[8];
    const float* pW1 = (const float*)d_in[9],  *pb1 = (const float*)d_in[10];
    const float* pW2 = (const float*)d_in[11], *pb2 = (const float*)d_in[12];
    const float* cW1 = (const float*)d_in[13], *cb1 = (const float*)d_in[14];
    const float* cW2 = (const float*)d_in[15], *cb2 = (const float*)d_in[16];
    const float* lW1 = (const float*)d_in[17], *lb1 = (const float*)d_in[18];
    const float* lW2 = (const float*)d_in[19], *lb2 = (const float*)d_in[20];
    const float* posW1 = (const float*)d_in[21], *posb1 = (const float*)d_in[22];
    const float* posW2 = (const float*)d_in[23], *posb2 = (const float*)d_in[24];
    const float* posW3 = (const float*)d_in[25], *posb3 = (const float*)d_in[26];
    const float* gW1 = (const float*)d_in[27], *gb1 = (const float*)d_in[28];
    const float* gW2 = (const float*)d_in[29], *gb2 = (const float*)d_in[30];
    const float* gW3 = (const float*)d_in[31], *gb3 = (const float*)d_in[32];

    const int E  = in_sizes[1] / 512;   // 65280
    const int Nn = in_sizes[3] / 4;     // 256

    // ----- workspace layout (floats, 256B-aligned chunks) -----
    float* wsf = (float*)d_ws;
    size_t off = 0;
    auto walloc = [&](size_t nElems) {
        float* p = wsf + off;
        off += (nElems + 63) & ~(size_t)63;
        return p;
    };
    float* posf  = walloc((size_t)E * 7);
    float* geomf = walloc((size_t)E * 4);
    float* t1    = walloc((size_t)E * 64);
    float* t2    = walloc((size_t)E * 64);
    float* posc  = walloc((size_t)E * 128);
    float* u1    = walloc((size_t)E * 32);
    float* u2    = walloc((size_t)E * 32);
    float* geomc = walloc((size_t)E * 64);
    float* h     = walloc((size_t)E * 512);

    float* outp     = (float*)d_out;
    float* link     = outp;
    float* support  = outp + (size_t)E * 512;
    float* prox     = outp + (size_t)2 * E * 512;
    float* comp     = outp + (size_t)3 * E * 512;

    const dim3 blk(256);

    edge_features_kernel<<<dim3((E + 255) / 256), blk, 0, stream>>>(
        centroid, lwhV, posf, geomf, Nn, E);

    auto gemm = [&](const float* A0, int K0, const float* A1, int K1,
                    const float* A2, int K2, const float* Wm, const float* b,
                    float* O, int N, int relu) {
        dim3 grid((E + TILE_M - 1) / TILE_M, (N + TILE_N - 1) / TILE_N);
        gemm_bias_act_kernel<<<grid, blk, 0, stream>>>(
            A0, A1, A2, K0, K1, K2, Wm, b, O, N, relu, E);
    };

    // pos FCN: 7 -> 64 -> 64 -> 128
    gemm(posf, 7,  nullptr, 0, nullptr, 0, posW1, posb1, t1,   64,  1);
    gemm(t1,   64, nullptr, 0, nullptr, 0, posW2, posb2, t2,   64,  1);
    gemm(t2,   64, nullptr, 0, nullptr, 0, posW3, posb3, posc, 128, 0);

    // geom FCN: 4 -> 32 -> 32 -> 64
    gemm(geomf, 4,  nullptr, 0, nullptr, 0, gW1, gb1, u1,    32, 1);
    gemm(u1,    32, nullptr, 0, nullptr, 0, gW2, gb2, u2,    32, 1);
    gemm(u2,    32, nullptr, 0, nullptr, 0, gW3, gb3, geomc, 64, 0);

    // support = MLP(pred)
    gemm(pred, 512, nullptr, 0, nullptr, 0, sW1, sb1, h,       512, 1);
    gemm(h,    512, nullptr, 0, nullptr, 0, sW2, sb2, support, 512, 0);

    // proximity = MLP(concat(pred, pos_codes))
    gemm(pred, 512, posc, 128, nullptr, 0, pW1, pb1, h,    512, 1);
    gemm(h,    512, nullptr, 0, nullptr, 0, pW2, pb2, prox, 512, 0);

    // comp = MLP(concat(pred, geom_codes))
    gemm(pred, 512, geomc, 64, nullptr, 0, cW1, cb1, h,    512, 1);
    gemm(h,    512, nullptr, 0, nullptr, 0, cW2, cb2, comp, 512, 0);

    // link = MLP(concat(support, proximity, comp))
    gemm(support, 512, prox, 512, comp, 512, lW1, lb1, h,    512, 1);
    gemm(h,       512, nullptr, 0, nullptr, 0, lW2, lb2, link, 512, 0);
}
